// SwitchMoE_53979148976156
// MI455X (gfx1250) — compile-verified, matrix operand order checked by reading, not run on hardware
//
#include <hip/hip_runtime.h>
#include <hip/hip_bf16.h>
#include <cstdint>

#define DIM  1024
#define HDIM 4096
#define NTOK 4096
#define NEXP 8
#define BM 128
#define BN 128
#define BK 32
#define LDA 40   // padded LDS row stride in halfs (80B, 16B-aligned)

typedef _Float16 h4  __attribute__((ext_vector_type(4)));
typedef _Float16 h8  __attribute__((ext_vector_type(8)));
typedef _Float16 h16 __attribute__((ext_vector_type(16)));
typedef float    v8f __attribute__((ext_vector_type(8)));

union H16 { h16 v; struct { h8 lo, hi; } p; };

// ---------------- gate kernels ----------------

__global__ void g0_zero(int* cnt) {
    int t = threadIdx.x;
    if (t < NEXP) cnt[t] = 0;
}

__global__ __launch_bounds__(256) void g1_gate(
    const float* __restrict__ x, const float* __restrict__ w_gate,
    const float* __restrict__ b_gate,
    int* __restrict__ cnt, int* __restrict__ top_idx,
    int* __restrict__ slot, float* __restrict__ top_g) {
    __shared__ float wg[DIM * NEXP];
    for (int i = threadIdx.x; i < DIM * NEXP; i += 256) wg[i] = w_gate[i];
    __syncthreads();

    int n = blockIdx.x * 256 + threadIdx.x;
    float acc[NEXP];
#pragma unroll
    for (int e = 0; e < NEXP; ++e) acc[e] = b_gate[e];

    const float4* x4 = (const float4*)(x + (size_t)n * DIM);
    for (int d4 = 0; d4 < DIM / 4; ++d4) {
        float4 xv = x4[d4];
        const float* wr = &wg[d4 * 4 * NEXP];
#pragma unroll
        for (int e = 0; e < NEXP; ++e) {
            acc[e] += xv.x * wr[e] + xv.y * wr[NEXP + e]
                    + xv.z * wr[2 * NEXP + e] + xv.w * wr[3 * NEXP + e];
        }
    }
    float m = acc[0]; int bi = 0;
#pragma unroll
    for (int e = 1; e < NEXP; ++e) if (acc[e] > m) { m = acc[e]; bi = e; }
    float s = 0.f;
#pragma unroll
    for (int e = 0; e < NEXP; ++e) s += __expf(acc[e] - m);
    float g = 1.0f / s;          // softmax value of the argmax entry
    top_idx[n] = bi;
    top_g[n]   = g;
    slot[n]    = atomicAdd(&cnt[bi], 1);   // int atomic: value-deterministic
}

// deterministic per-expert column sum of masked gate scores
__global__ __launch_bounds__(256) void g1b_colsum(
    const int* __restrict__ top_idx, const float* __restrict__ top_g,
    float* __restrict__ colsum) {
    __shared__ float red[256];
    int e = blockIdx.x;
    float s = 0.f;
    for (int n = threadIdx.x; n < NTOK; n += 256)
        s += (top_idx[n] == e) ? top_g[n] : 0.f;
    red[threadIdx.x] = s;
    __syncthreads();
    for (int o = 128; o > 0; o >>= 1) {
        if (threadIdx.x < o) red[threadIdx.x] += red[threadIdx.x + o];
        __syncthreads();
    }
    if (threadIdx.x == 0) colsum[e] = red[0];
}

__global__ void g2_finalize(const float* __restrict__ colsum,
                            const int* __restrict__ cnt,
                            int* __restrict__ base, float* __restrict__ scale,
                            float* __restrict__ loss_out) {
    if (threadIdx.x == 0) {
        float load[NEXP]; int b = 0; float tot = 0.f;
        for (int e = 0; e < NEXP; ++e) {
            base[e] = b; b += cnt[e];
            float sc = (float)NTOK / (colsum[e] + 1e-6f);
            scale[e] = sc;
            load[e] = colsum[e] * sc;
            tot += load[e];
        }
        float imp = tot / (float)NTOK;
        float l = 0.f;
        for (int e = 0; e < NEXP; ++e) { float d = load[e] - imp; l += d * d; }
        *loss_out = l / (float)NEXP;
    }
}

__global__ __launch_bounds__(256) void g3_scatter(
    const int* __restrict__ top_idx, const int* __restrict__ slot,
    const float* __restrict__ top_g, const int* __restrict__ base,
    const float* __restrict__ scale, int* __restrict__ list,
    float* __restrict__ gv) {
    int n = blockIdx.x * 256 + threadIdx.x;
    int e = top_idx[n];
    list[base[e] + slot[n]] = n;
    gv[n] = top_g[n] * scale[e];
}

// ---------------- WMMA tile compute (shared by both GEMMs) ----------------

__device__ __forceinline__ void mma_tile(const _Float16* ldsA,
                                         const _Float16* ldsB,
                                         int m0w, int n0w, int l15, int khalf,
                                         v8f acc[4][2]) {
    H16 bf[2];
#pragma unroll
    for (int j = 0; j < 2; ++j) {
        int nr = n0w + j * 16 + l15;
        bf[j].p.lo = *(const h8*)&ldsB[nr * LDA + khalf];
        bf[j].p.hi = *(const h8*)&ldsB[nr * LDA + khalf + 16];
    }
#pragma unroll
    for (int i = 0; i < 4; ++i) {
        H16 af;
        int mr = m0w + i * 16 + l15;
        af.p.lo = *(const h8*)&ldsA[mr * LDA + khalf];
        af.p.hi = *(const h8*)&ldsA[mr * LDA + khalf + 16];
#pragma unroll
        for (int j = 0; j < 2; ++j)
            acc[i][j] = __builtin_amdgcn_wmma_f32_16x16x32_f16(
                false, af.v, false, bf[j].v, (short)0, acc[i][j], false, false);
    }
}

__device__ __forceinline__ void store_b16(float bv[16], _Float16* dstRow) {
    h8 lo = {(_Float16)bv[0], (_Float16)bv[1], (_Float16)bv[2], (_Float16)bv[3],
             (_Float16)bv[4], (_Float16)bv[5], (_Float16)bv[6], (_Float16)bv[7]};
    h8 hi = {(_Float16)bv[8], (_Float16)bv[9], (_Float16)bv[10], (_Float16)bv[11],
             (_Float16)bv[12], (_Float16)bv[13], (_Float16)bv[14], (_Float16)bv[15]};
    *(h8*)(dstRow)     = lo;
    *(h8*)(dstRow + 8) = hi;
}

// ---------------- expert GEMMs ----------------

// H[perm(n), :] = gelu(x[n,:] @ w1[e] + b1[e]) for tokens routed to expert e
__global__ __launch_bounds__(256) void gemm1_gelu(
    const float* __restrict__ x, const float* __restrict__ w1,
    const float* __restrict__ b1, const int* __restrict__ list,
    const int* __restrict__ cnt, const int* __restrict__ base,
    _Float16* __restrict__ Hbuf) {
    int e = blockIdx.z;
    int cnt_e = cnt[e];
    int m0blk = blockIdx.y * BM;
    if (m0blk >= cnt_e) return;
    int n0blk = blockIdx.x * BN;
    int base_e = base[e];
    const float* w1e = w1 + (size_t)e * DIM * HDIM;

    __shared__ _Float16 ldsA[2][BM * LDA];
    __shared__ _Float16 ldsB[2][BN * LDA];

    int tid = threadIdx.x;
    int lane = tid & 31;
    int wv = tid >> 5;
    int m0w = (wv >> 2) * 64;    // wave M origin (2 waves in M)
    int n0w = (wv & 3) * 32;     // wave N origin (4 waves in N)
    int l15 = lane & 15;
    int khalf = (lane >> 4) * 8;

    // A staging: 2 threads per row, 16 f32 each; row CLAMPED (no predication,
    // duplicate rows are never stored in the epilogue)
    int arow = tid >> 1;
    int aseg = (tid & 1) * 16;
    int gm_a = m0blk + arow;
    int gm_c = gm_a < cnt_e ? gm_a : cnt_e - 1;
    int tok_a = list[base_e + gm_c];
    const float* asrc = x + (size_t)tok_a * DIM + aseg;

    // B staging: thread owns one n-column, 16 consecutive k (coalesced loads,
    // 2x ds_store_b128 into transposed [N][K] LDS layout)
    int bn  = tid & 127;
    int bk0 = (tid >> 7) * 16;
    const float* bptr = w1e + (size_t)bk0 * HDIM + n0blk + bn;

    v8f acc[4][2] = {};

    const int T = DIM / BK;
    // prologue: stage tile 0
    {
        float4 av[4];
#pragma unroll
        for (int q = 0; q < 4; ++q) av[q] = *(const float4*)(asrc + q * 4);
        store_b16((float*)av, &ldsA[0][arow * LDA + aseg]);
        float bv[16];
#pragma unroll
        for (int k = 0; k < 16; ++k) bv[k] = bptr[(size_t)k * HDIM];
        store_b16(bv, &ldsB[0][bn * LDA + bk0]);
    }
    __syncthreads();

    for (int t = 0; t < T; ++t) {
        int kn = (t + 1) * BK;
        float4 av[4]; float bv[16];
        if (t + 1 < T) {
#pragma unroll
            for (int q = 0; q < 4; ++q) av[q] = *(const float4*)(asrc + kn + q * 4);
#pragma unroll
            for (int k = 0; k < 16; ++k) bv[k] = bptr[(size_t)(kn + k) * HDIM];
            if (t + 2 < T) {
                __builtin_prefetch(bptr + (size_t)(kn + BK) * HDIM, 0, 1);
                __builtin_prefetch(asrc + kn + BK, 0, 1);
            }
        }
        mma_tile(ldsA[t & 1], ldsB[t & 1], m0w, n0w, l15, khalf, acc);
        if (t + 1 < T) {
            store_b16((float*)av, &ldsA[(t + 1) & 1][arow * LDA + aseg]);
            store_b16(bv, &ldsB[(t + 1) & 1][bn * LDA + bk0]);
            __syncthreads();
        }
    }

    // epilogue: bias + fast-gelu, store f16 into packed H
#pragma unroll
    for (int i = 0; i < 4; ++i) {
#pragma unroll
        for (int j = 0; j < 2; ++j) {
            int col = n0blk + n0w + j * 16 + l15;
            float bias = b1[(size_t)e * HDIM + col];
#pragma unroll
            for (int r = 0; r < 8; ++r) {
                int mrow = m0w + i * 16 + (lane >> 4) * 8 + r;
                int gm = m0blk + mrow;
                if (gm < cnt_e) {
                    float y = acc[i][j][r] + bias;
                    float sg = 1.0f / (1.0f + __expf(-1.702f * y));
                    Hbuf[(size_t)(base_e + gm) * HDIM + col] = (_Float16)(y * sg);
                }
            }
        }
    }
}

// out[token,:] = gv[token] * (H[perm(token),:] @ w2[e] + b2[e])
__global__ __launch_bounds__(256) void gemm2_combine(
    const _Float16* __restrict__ Hbuf, const float* __restrict__ w2,
    const float* __restrict__ b2, const int* __restrict__ list,
    const int* __restrict__ cnt, const int* __restrict__ base,
    const float* __restrict__ gv, float* __restrict__ out) {
    int e = blockIdx.z;
    int cnt_e = cnt[e];
    int m0blk = blockIdx.y * BM;
    if (m0blk >= cnt_e) return;
    int n0blk = blockIdx.x * BN;
    int base_e = base[e];
    const float* w2e = w2 + (size_t)e * HDIM * DIM;

    __shared__ _Float16 ldsA[2][BM * LDA];
    __shared__ _Float16 ldsB[2][BN * LDA];

    int tid = threadIdx.x;
    int lane = tid & 31;
    int wv = tid >> 5;
    int m0w = (wv >> 2) * 64;
    int n0w = (wv & 3) * 32;
    int l15 = lane & 15;
    int khalf = (lane >> 4) * 8;

    int arow = tid >> 1;
    int aseg = (tid & 1) * 16;
    int gm_a = m0blk + arow;
    int gm_c = gm_a < cnt_e ? gm_a : cnt_e - 1;   // clamp, no predication
    const _Float16* arow_ptr = Hbuf + (size_t)(base_e + gm_c) * HDIM + aseg;

    int bn  = tid & 127;
    int bk0 = (tid >> 7) * 16;
    const float* bptr = w2e + (size_t)bk0 * DIM + n0blk + bn;

    v8f acc[4][2] = {};

    const int T = HDIM / BK;
    {
        h8 a0 = *(const h8*)(arow_ptr);
        h8 a1 = *(const h8*)(arow_ptr + 8);
        *(h8*)&ldsA[0][arow * LDA + aseg]     = a0;
        *(h8*)&ldsA[0][arow * LDA + aseg + 8] = a1;
        float bv[16];
#pragma unroll
        for (int k = 0; k < 16; ++k) bv[k] = bptr[(size_t)k * DIM];
        store_b16(bv, &ldsB[0][bn * LDA + bk0]);
    }
    __syncthreads();

    for (int t = 0; t < T; ++t) {
        int kn = (t + 1) * BK;
        h8 a0, a1; float bv[16];
        if (t + 1 < T) {
            a0 = *(const h8*)(arow_ptr + kn);
            a1 = *(const h8*)(arow_ptr + kn + 8);
#pragma unroll
            for (int k = 0; k < 16; ++k) bv[k] = bptr[(size_t)(kn + k) * DIM];
            if (t + 2 < T) {
                __builtin_prefetch(bptr + (size_t)(kn + BK) * DIM, 0, 1);
                __builtin_prefetch(arow_ptr + kn + BK, 0, 1);
            }
        }
        mma_tile(ldsA[t & 1], ldsB[t & 1], m0w, n0w, l15, khalf, acc);
        if (t + 1 < T) {
            *(h8*)&ldsA[(t + 1) & 1][arow * LDA + aseg]     = a0;
            *(h8*)&ldsA[(t + 1) & 1][arow * LDA + aseg + 8] = a1;
            store_b16(bv, &ldsB[(t + 1) & 1][bn * LDA + bk0]);
            __syncthreads();
        }
    }

#pragma unroll
    for (int i = 0; i < 4; ++i) {
#pragma unroll
        for (int j = 0; j < 2; ++j) {
            int col = n0blk + n0w + j * 16 + l15;
            float bias = b2[(size_t)e * DIM + col];
#pragma unroll
            for (int r = 0; r < 8; ++r) {
                int mrow = m0w + i * 16 + (lane >> 4) * 8 + r;
                int gm = m0blk + mrow;
                if (gm < cnt_e) {
                    int tok = list[base_e + gm];
                    float y = acc[i][j][r] + bias;
                    out[(size_t)tok * DIM + col] = gv[tok] * y;
                }
            }
        }
    }
}

// ---------------- launcher ----------------

extern "C" void kernel_launch(void* const* d_in, const int* in_sizes, int n_in,
                              void* d_out, int out_size, void* d_ws, size_t ws_size,
                              hipStream_t stream) {
    (void)in_sizes; (void)n_in; (void)out_size; (void)ws_size;
    const float* x      = (const float*)d_in[0];
    const float* w_gate = (const float*)d_in[1];
    const float* b_gate = (const float*)d_in[2];
    const float* w1     = (const float*)d_in[3];
    const float* b1     = (const float*)d_in[4];
    const float* w2     = (const float*)d_in[5];
    const float* b2     = (const float*)d_in[6];
    float* out = (float*)d_out;

    char* w = (char*)d_ws;
    float* colsum  = (float*)(w + 0);
    int*   cnt     = (int*)  (w + 32);
    int*   base    = (int*)  (w + 64);
    float* scale   = (float*)(w + 96);
    int*   top_idx = (int*)  (w + 128);
    int*   slot    = (int*)  (w + 128 + 1 * 16384);
    float* top_g   = (float*)(w + 128 + 2 * 16384);
    float* gv      = (float*)(w + 128 + 3 * 16384);
    int*   list    = (int*)  (w + 128 + 4 * 16384);
    _Float16* Hbuf = (_Float16*)(w + 131072);   // 4096*4096 f16 = 32 MiB

    g0_zero<<<1, 32, 0, stream>>>(cnt);
    g1_gate<<<NTOK / 256, 256, 0, stream>>>(x, w_gate, b_gate, cnt, top_idx, slot, top_g);
    g1b_colsum<<<NEXP, 256, 0, stream>>>(top_idx, top_g, colsum);
    g2_finalize<<<1, 32, 0, stream>>>(colsum, cnt, base, scale, out + (size_t)NTOK * DIM);
    g3_scatter<<<NTOK / 256, 256, 0, stream>>>(top_idx, slot, top_g, base, scale, list, gv);
    gemm1_gelu<<<dim3(HDIM / BN, NTOK / BM, NEXP), 256, 0, stream>>>(
        x, w1, b1, list, cnt, base, Hbuf);
    gemm2_combine<<<dim3(DIM / BN, NTOK / BM, NEXP), 256, 0, stream>>>(
        Hbuf, w2, b2, list, cnt, base, gv, out);
}
